// LearntNeighbourhoodSampling_19198503813327
// MI455X (gfx1250) — matrix operand order repaired and checked by reading, not compiled, not run on hardware
//
#include <hip/hip_runtime.h>
#include <hip/hip_bf16.h>
#include <stdint.h>

#define DI __device__ __forceinline__

constexpr int cB = 2, cC = 128, cH = 800, cW = 640, cN = 8192;
constexpr int C2 = 130;            // C + 2 coord channels
constexpr int KTOT = 9 * C2;       // 1170
constexpr int KP = 1184;           // padded reduction dim (37 * 32)
constexpr int KTILES = KP / 32;    // 37
constexpr int R = cB * cN;         // 16384 rows
constexpr int OTILES = cC / 16;    // 8 output-channel tiles

typedef __bf16   v16bf __attribute__((ext_vector_type(16)));
typedef float    v8f   __attribute__((ext_vector_type(8)));
typedef uint32_t v4u   __attribute__((ext_vector_type(4)));

DI uint16_t bf16rne(float f) {
    uint32_t u = __float_as_uint(f);
    uint32_t r = u + 0x7FFFu + ((u >> 16) & 1u);
    return (uint16_t)(r >> 16);
}

// Async DMA 16B from global memory straight into LDS (no VGPR round-trip).
// LDS byte address is taken from vdst; low 32 bits of a generic shared pointer are
// the allocation-relative LDS byte offset. Tracked by ASYNCcnt.
DI void async_copy_b128(const void* g, void* l) {
    uint32_t loff = (uint32_t)(uintptr_t)l;
    asm volatile("global_load_async_to_lds_b128 %0, %1, off"
                 :: "v"(loff), "v"((uint64_t)(uintptr_t)g) : "memory");
}
DI void wait_async0() {
    asm volatile("s_wait_asynccnt 0x0" ::: "memory");
}

// ---------------- NCHW -> NHWC transpose (coalesced both ways via LDS) ----------
__global__ __launch_bounds__(256) void ktranspose(const float* __restrict__ in,
                                                  float* __restrict__ out) {
    __shared__ float tile[32][33];
    int bh = blockIdx.z;                 // b*H + y
    int x0 = blockIdx.x * 32, c0 = blockIdx.y * 32;
    int tx = threadIdx.x, ty = threadIdx.y;   // 32 x 8
    const size_t HW = (size_t)cH * cW;
    int b = bh / cH, y = bh % cH;
    for (int i = 0; i < 32; i += 8)
        tile[ty + i][tx] = in[((size_t)(b * cC + c0 + ty + i)) * HW + (size_t)y * cW + x0 + tx];
    __syncthreads();
    for (int i = 0; i < 32; i += 8)
        out[((size_t)bh * cW + x0 + ty + i) * cC + c0 + tx] = tile[tx][ty + i];
}

// ---------------- bilinear sampler (border clamp, align_corners=True) -----------
DI float bilin(const float* __restrict__ img, int b, int c, float gx, float gy, int nhwc) {
    float x = (gx + 1.0f) * 0.5f * (float)(cW - 1);
    float y = (gy + 1.0f) * 0.5f * (float)(cH - 1);
    x = fminf(fmaxf(x, 0.0f), (float)(cW - 1));
    y = fminf(fmaxf(y, 0.0f), (float)(cH - 1));
    float xf = floorf(x), yf = floorf(y);
    int x0 = (int)xf, y0 = (int)yf;
    int x1 = (x0 + 1 < cW) ? x0 + 1 : cW - 1;
    int y1 = (y0 + 1 < cH) ? y0 + 1 : cH - 1;
    float wx = x - xf, wy = y - yf;
    float v00, v01, v10, v11;
    if (nhwc) {
        size_t r0 = (size_t)(b * cH + y0) * cW;
        size_t r1 = (size_t)(b * cH + y1) * cW;
        v00 = img[(r0 + x0) * cC + c]; v01 = img[(r0 + x1) * cC + c];
        v10 = img[(r1 + x0) * cC + c]; v11 = img[(r1 + x1) * cC + c];
    } else {
        size_t pb = ((size_t)b * cC + c) * cH;
        v00 = img[(pb + y0) * cW + x0]; v01 = img[(pb + y0) * cW + x1];
        v10 = img[(pb + y1) * cW + x0]; v11 = img[(pb + y1) * cW + x1];
    }
    return v00 * (1 - wx) * (1 - wy) + v01 * wx * (1 - wy)
         + v10 * (1 - wx) * wy + v11 * wx * wy;
}

// ---------------- fuse Wd2@Wd1, Wc2@Wc1 and biases ------------------------------
__global__ __launch_bounds__(256) void kfuse1(
    const float* Wd1, const float* bd1, const float* Wd2, const float* bd2,
    const float* Wc1, const float* bc1, const float* Wc2, const float* bc2,
    float* Wdd, float* Wcc, float* bdd, float* bcc) {
    int idx = blockIdx.x * 256 + threadIdx.x;
    if (idx < cC * C2) {
        int o = idx / C2, j = idx % C2; float a = 0.f;
        for (int c = 0; c < cC; c++) a += Wd2[o * cC + c] * Wd1[c * C2 + j];
        Wdd[idx] = a;
    } else if (idx < 2 * cC * C2) {
        int t = idx - cC * C2; int o = t / C2, j = t % C2; float a = 0.f;
        for (int c = 0; c < cC; c++) a += Wc2[o * cC + c] * Wc1[c * C2 + j];
        Wcc[t] = a;
    } else if (idx < 2 * cC * C2 + cC) {
        int o = idx - 2 * cC * C2; float a = bd2[o];
        for (int c = 0; c < cC; c++) a += Wd2[o * cC + c] * bd1[c];
        bdd[o] = a;
    } else if (idx < 2 * cC * C2 + 2 * cC) {
        int o = idx - (2 * cC * C2 + cC); float a = bc2[o];
        for (int c = 0; c < cC; c++) a += Wc2[o * cC + c] * bc1[c];
        bcc[o] = a;
    }
}

// ---------------- Meff[o, k*130+j] = sum_c Wsn[o,c,k] * Wdd[c,j] (f32, padded) --
__global__ __launch_bounds__(256) void kMeff(const float* __restrict__ Wsn,
                                             const float* __restrict__ Wdd,
                                             float* __restrict__ Meff) {
    int idx = blockIdx.x * 256 + threadIdx.x;
    if (idx >= cC * KP) return;
    int o = idx / KP, col = idx % KP;
    float a = 0.f;
    if (col < KTOT) {
        int k = col / C2, j = col % C2;
        for (int c = 0; c < cC; c++) a += Wsn[(o * cC + c) * 9 + k] * Wdd[c * C2 + j];
    }
    Meff[idx] = a;
}

// ---------------- fold center path + diff-subtraction into the k=0 slab ---------
__global__ __launch_bounds__(256) void kAdjust(
    const float* __restrict__ Wcc, const float* __restrict__ Wsn,
    const float* __restrict__ bdd, const float* __restrict__ bsn,
    const float* __restrict__ bcc, float* __restrict__ Meff,
    float* __restrict__ beff) {
    int idx = blockIdx.x * 256 + threadIdx.x;
    if (idx >= cC * C2) return;
    int o = idx / C2, j = idx % C2;
    float S = 0.f;
    for (int k = 0; k < 9; k++) S += Meff[o * KP + k * C2 + j];
    float m0 = Meff[o * KP + j];
    Meff[o * KP + j] = m0 - S + Wcc[o * C2 + j];
    if (j == 0) {
        float a = bsn[o] + bcc[o];
        for (int c = 0; c < cC; c++) {
            float s = 0.f;
            for (int k = 0; k < 9; k++) s += Wsn[(o * cC + c) * 9 + k];
            a += bdd[c] * s;
        }
        beff[o] = a;
    }
}

// ---------------- pack Meff into WMMA B-matrix (32x16 bf16) lanes, contiguous k = 16*(lane/16) + s
__global__ __launch_bounds__(256) void kPackB(const float* __restrict__ Meff,
                                              uint32_t* __restrict__ Bbuf) {
    int idx = blockIdx.x * 256 + threadIdx.x;
    if (idx >= OTILES * KTILES * 32 * 8) return;
    int v = idx % 8;
    int rest = idx / 8;
    int lane = rest % 32;
    int rest2 = rest / 32;
    int kt = rest2 % KTILES;
    int ot = rest2 / KTILES;
    int o  = ot * 16 + (lane & 15);
    int k0 = (lane & 16) ? 16 : 0;
    int kk = kt * 32 + k0 + 2 * v;
    uint32_t lo = bf16rne(Meff[o * KP + kk]);
    uint32_t hi = bf16rne(Meff[o * KP + kk + 1]);
    Bbuf[idx] = lo | (hi << 16);
}

// ---------------- sample 128 channels at vertex centers -------------------------
__global__ __launch_bounds__(128) void kCenter(const float* __restrict__ img,
                                               const float* __restrict__ verts,
                                               float* __restrict__ featc, int nhwc) {
    int p = blockIdx.x, c = threadIdx.x;
    float gx = verts[p * 2], gy = verts[p * 2 + 1];
    featc[(size_t)p * cC + c] = bilin(img, p >> 13, c, gx, gy, nhwc);
}

// ---------------- per-vertex 18-dim offset head; build neighbour coords ---------
__global__ __launch_bounds__(256) void kOffsets(const float* __restrict__ Wsd,
                                                const float* __restrict__ bsd,
                                                const float* __restrict__ featc,
                                                const float* __restrict__ verts,
                                                float* __restrict__ nbuf) {
    int idx = blockIdx.x * 256 + threadIdx.x;
    if (idx >= R * 18) return;
    int p = idx / 18, o = idx % 18;
    int k = o >> 1, comp = o & 1;
    float v = verts[p * 2 + comp];
    if (k > 0) {                          // offset 0 forced to zero (center)
        float a = bsd[o];
        const float* f = featc + (size_t)p * cC;
        for (int c = 0; c < cC; c++) a += Wsd[o * cC + c] * f[c];
        v += a;
    }
    nbuf[(p * 9 + k) * 2 + comp] = v;
}

// ---------------- A-matrix: k=0 slab (reuse center samples) + zero pad ----------
__global__ __launch_bounds__(256) void kFillK0(const float* __restrict__ featc,
                                               const float* __restrict__ verts,
                                               uint16_t* __restrict__ Abuf) {
    int p = blockIdx.x;
    uint16_t* arow = Abuf + (size_t)p * KP;
    for (int col = threadIdx.x; col < KP; col += 256) {
        if (col < cC)        arow[col] = bf16rne(featc[(size_t)p * cC + col]);
        else if (col < C2)   arow[col] = bf16rne(verts[p * 2 + (col - cC)]);
        else if (col >= KTOT) arow[col] = 0;
    }
}

// ---------------- A-matrix: neighbour slabs k=1..8 (the big gather) -------------
__global__ __launch_bounds__(160) void kNb(const float* __restrict__ img,
                                           const float* __restrict__ nbuf,
                                           uint16_t* __restrict__ Abuf, int nhwc) {
    int k = blockIdx.x + 1;
    int p = blockIdx.y;
    int c = threadIdx.x;
    if (c >= C2) return;
    float gx = nbuf[(p * 9 + k) * 2], gy = nbuf[(p * 9 + k) * 2 + 1];
    float val;
    if (c < cC) val = bilin(img, p >> 13, c, gx, gy, nhwc);
    else        val = (c == cC) ? gx : gy;
    Abuf[(size_t)p * KP + k * C2 + c] = bf16rne(val);
}

// ---------------- fused tail: single bf16 WMMA GEMM 16384 x 1184 x 128 ----------
// Block = 8 waves, 32 rows x 128 cols. Wave w owns o-tile w with two M-tiles
// (two accumulators) so each B fragment feeds two back-to-back WMMAs.
// A tile (32 x 1184 bf16, 76.8 KB) staged via async global->LDS DMA.
__global__ __launch_bounds__(256) void kGemm(const uint16_t* __restrict__ Abuf,
                                             const uint32_t* __restrict__ Bbuf,
                                             const float* __restrict__ beff,
                                             float* __restrict__ out) {
    extern __shared__ __align__(16) uint16_t smem[];     // 32 rows, stride 2400 B
    int tid = threadIdx.x;
    const uint16_t* Atile = Abuf + (size_t)blockIdx.x * (32 * KP);
    for (int i = tid; i < 32 * 148; i += 256) {          // 148 x 16B per row
        int row = i / 148, off = i % 148;
        async_copy_b128(Atile + row * KP + off * 8, (char*)smem + row * 2400 + off * 16);
    }
    wait_async0();
    __syncthreads();

    int lane = tid & 31;
    int wave = tid >> 5;                 // output-channel tile 0..7
    int hi16 = (lane >> 4) & 1;          // K-half select per A/B lane layouts
    union U { v4u q[2]; v16bf v; };
    v8f acc0 = {0.f, 0.f, 0.f, 0.f, 0.f, 0.f, 0.f, 0.f};
    v8f acc1 = acc0;
    const char* a0 = (const char*)smem + (lane & 15) * 2400 + hi16 * 32;
    const char* a1 = a0 + 16 * 2400;

    for (int kt = 0; kt < KTILES; kt++) {
        U b, am0, am1;
        const v4u* bp = (const v4u*)(Bbuf + (((size_t)wave * KTILES + kt) * 32 + lane) * 8);
        b.q[0]   = bp[0];
        b.q[1]   = bp[1];
        am0.q[0] = *(const v4u*)(a0 + kt * 64);          // K = k0..k0+7
        am0.q[1] = *(const v4u*)(a0 + kt * 64 + 16);     // K = k0+8..k0+15
        am1.q[0] = *(const v4u*)(a1 + kt * 64);
        am1.q[1] = *(const v4u*)(a1 + kt * 64 + 16);
        acc0 = __builtin_amdgcn_wmma_f32_16x16x32_bf16(
            false, am0.v, false, b.v, (short)0, acc0, false, false);
        acc1 = __builtin_amdgcn_wmma_f32_16x16x32_bf16(
            false, am1.v, false, b.v, (short)0, acc1, false, false);
    }

    int ncol  = wave * 16 + (lane & 15);
    int mbase = blockIdx.x * 32 + hi16 * 8;
    float bo = beff[ncol];
    for (int r = 0; r < 8; r++) {
        out[(size_t)(mbase + r) * cC + ncol]      = acc0[r] + bo;
        out[(size_t)(mbase + 16 + r) * cC + ncol] = acc1[r] + bo;
    }
}

// --------------------------------------------------------------------------------
extern "C" void kernel_launch(void* const* d_in, const int* in_sizes, int n_in,
                              void* d_out, int out_size, void* d_ws, size_t ws_size,
                              hipStream_t stream) {
    const float* img   = (const float*)d_in[0];
    const float* verts = (const float*)d_in[1];
    const float* Wsd   = (const float*)d_in[2];
    const float* bsd   = (const float*)d_in[3];
    const float* Wd1   = (const float*)d_in[4];
    const float* bd1   = (const float*)d_in[5];
    const float* Wd2   = (const float*)d_in[6];
    const float* bd2   = (const float*)d_in[7];
    const float* Wsn   = (const float*)d_in[8];
    const float* bsn   = (const float*)d_in[9];
    const float* Wc1   = (const float*)d_in[10];
    const float* bc1   = (const float*)d_in[11];
    const float* Wc2   = (const float*)d_in[12];
    const float* bc2   = (const float*)d_in[13];
    float* out = (float*)d_out;

    char* ws = (char*)d_ws;
    size_t off = 0;
    auto alloc = [&](size_t bytes) { size_t o = off; off += (bytes + 255) & ~(size_t)255; return o; };
    float*    Wdd   = (float*)(ws + alloc((size_t)cC * C2 * 4));
    float*    Wcc   = (float*)(ws + alloc((size_t)cC * C2 * 4));
    float*    bdd   = (float*)(ws + alloc(512));
    float*    bcc   = (float*)(ws + alloc(512));
    float*    Meff  = (float*)(ws + alloc((size_t)cC * KP * 4));
    float*    beff  = (float*)(ws + alloc(512));
    float*    featc = (float*)(ws + alloc((size_t)R * cC * 4));
    float*    nbuf  = (float*)(ws + alloc((size_t)R * 18 * 4));
    uint16_t* Abuf  = (uint16_t*)(ws + alloc((size_t)R * KP * 2));
    uint32_t* Bbuf  = (uint32_t*)(ws + alloc((size_t)OTILES * KTILES * 32 * 8 * 4));

    // NHWC copy of the image makes every bilinear tap a coalesced 512B channel run.
    size_t nhwc_off   = off;
    size_t nhwc_bytes = (size_t)cB * cC * cH * cW * 4;
    int use_nhwc = (nhwc_off + nhwc_bytes <= ws_size) ? 1 : 0;
    const float* simg = img;
    if (use_nhwc) {
        float* nh = (float*)(ws + nhwc_off);
        ktranspose<<<dim3(cW / 32, cC / 32, cB * cH), dim3(32, 8), 0, stream>>>(img, nh);
        simg = nh;
    }

    kfuse1<<<(2 * cC * C2 + 2 * cC + 255) / 256, 256, 0, stream>>>(
        Wd1, bd1, Wd2, bd2, Wc1, bc1, Wc2, bc2, Wdd, Wcc, bdd, bcc);
    kMeff<<<(cC * KP + 255) / 256, 256, 0, stream>>>(Wsn, Wdd, Meff);
    kAdjust<<<(cC * C2 + 255) / 256, 256, 0, stream>>>(Wcc, Wsn, bdd, bsn, bcc, Meff, beff);
    kPackB<<<(OTILES * KTILES * 32 * 8 + 255) / 256, 256, 0, stream>>>(Meff, Bbuf);

    kCenter<<<R, cC, 0, stream>>>(simg, verts, featc, use_nhwc);
    kOffsets<<<(R * 18 + 255) / 256, 256, 0, stream>>>(Wsd, bsd, featc, verts, nbuf);
    kFillK0<<<R, 256, 0, stream>>>(featc, verts, Abuf);
    kNb<<<dim3(8, R), 160, 0, stream>>>(simg, nbuf, Abuf, use_nhwc);

    kGemm<<<R / 32, 256, 32 * 2400, stream>>>(Abuf, Bbuf, beff, out);
}
